// MyRNN_81956565942831
// MI455X (gfx1250) — matrix operand order repaired and checked by reading, not compiled, not run on hardware
//
#include <hip/hip_runtime.h>
#include <hip/hip_bf16.h>
#include <math.h>

#define B_   1024
#define T_   80
#define V_   50000
#define E_   512
#define U_   512
#define STR  528   // LDS row stride in bf16 elements (512 + 16 pad to skew banks)
#define NBT  (B_ / 16)   // 64 B-tiles

typedef __attribute__((ext_vector_type(16))) __bf16 v16bf;
typedef __attribute__((ext_vector_type(8)))  float  v8f;

static __device__ __forceinline__ unsigned short f2bf(float f) {
    union { float f; unsigned int u; } c; c.f = f;
    unsigned int u = c.u;
    return (unsigned short)((u + 0x7FFFu + ((u >> 16) & 1u)) >> 16);  // RNE
}
static __device__ __forceinline__ float bf2f(unsigned short h) {
    union { unsigned int u; float f; } c; c.u = ((unsigned int)h) << 16;
    return c.f;
}
static __device__ __forceinline__ float fast_tanh(float x) {
#if __has_builtin(__builtin_amdgcn_tanhf)
    return __builtin_amdgcn_tanhf(x);
#else
    return tanhf(x);
#endif
}

union FragBF { uint4 q[2]; v16bf v; };
union H8     { uint4 q; unsigned short h[8]; };

// B-fragment (pre-shuffled layout): fragment f occupies 1024 bytes,
// lane holds 16 contiguous bf16 (32B). half h -> K = kt*32 + (lane>=16)*8 + h + (h>=8)*8,
// N = nt*16 + (lane&15), with f = kt*32 + nt.
static __device__ __forceinline__ v16bf load_bfrag(const unsigned short* __restrict__ base,
                                                   int f, int lane) {
    const uint4* p = (const uint4*)(base + (size_t)f * 512 + lane * 16);
    FragBF fr; fr.q[0] = p[0]; fr.q[1] = p[1];
    return fr.v;
}

// A-fragment from LDS tile [16 rows][STR], row = lane&15.
// ISA A layout: lanes<16 -> K = {kt*32+0..7, kt*32+16..23}; lanes>=16 -> +8.
static __device__ __forceinline__ v16bf load_afrag_lds(const unsigned short* Alds,
                                                       int kt, int lane) {
    const unsigned short* p = Alds + (lane & 15) * STR + kt * 32 + ((lane >> 4) << 3);
    FragBF fr;
    fr.q[0] = *(const uint4*)(p);
    fr.q[1] = *(const uint4*)(p + 16);
    return fr.v;
}

// A-fragment straight from a global bf16 row (row base chosen per lane).
static __device__ __forceinline__ v16bf load_afrag_g(const unsigned short* __restrict__ p) {
    FragBF fr;
    fr.q[0] = *(const uint4*)(p);
    fr.q[1] = *(const uint4*)(p + 16);
    return fr.v;
}

// ---------------------------------------------------------------------------
// Kernel 0a: fp32 [512 x 512] (K x N, row-major) -> bf16 B-fragment layout
// ---------------------------------------------------------------------------
__global__ __launch_bounds__(32)
void shuffleB_kernel(const float* __restrict__ src, unsigned short* __restrict__ dst) {
    int f    = blockIdx.x;      // 0..511  (kt*32 + nt)
    int lane = threadIdx.x;     // 0..31
    int kt = f >> 5, nt = f & 31;
    int n     = nt * 16 + (lane & 15);
    int kbase = kt * 32 + ((lane >> 4) << 3);
    unsigned short tmp[16];
#pragma unroll
    for (int h = 0; h < 16; ++h)
        tmp[h] = f2bf(src[(size_t)(kbase + h + ((h >> 3) << 3)) * U_ + n]);
    uint4* out = (uint4*)(dst + (size_t)f * 512 + lane * 16);
    out[0] = *(uint4*)(&tmp[0]);
    out[1] = *(uint4*)(&tmp[8]);
}

// ---------------------------------------------------------------------------
// Kernel 0b: emb fp32 -> bf16 (halves gather traffic; enables direct A frags)
// ---------------------------------------------------------------------------
__global__ __launch_bounds__(256)
void cvt_emb_kernel(const float* __restrict__ src, unsigned short* __restrict__ dst, int n4) {
    int i = blockIdx.x * 256 + threadIdx.x;
    if (i >= n4) return;
    float4 v = ((const float4*)src)[i];
    unsigned short h4[4] = { f2bf(v.x), f2bf(v.y), f2bf(v.z), f2bf(v.w) };
    ((uint2*)dst)[i] = *(const uint2*)h4;
}

// ---------------------------------------------------------------------------
// Kernel 1: xk = emb_bf[tokens] @ kernel + bias, stored bf16 in C-fragment
// layout: frag fi = ((t*NBT + bt)*4 + wn)*8 + nt ; 256 halves/frag (lane*8+r).
// WG = 16 B-rows x full N=512; 4 waves; wave wn owns cols [wn*128,+128).
// ---------------------------------------------------------------------------
__global__ __launch_bounds__(128)
void proj_kernel(const int* __restrict__ tokens, const unsigned short* __restrict__ emb_bf,
                 const unsigned short* __restrict__ Wshuf,
                 const float* __restrict__ bias, unsigned short* __restrict__ xk) {
    __shared__ int toks[16];
    const int bt  = blockIdx.x;          // B tile (16 rows)
    const int t   = blockIdx.y;
    const int tid = threadIdx.x;
    const int lane = tid & 31;
    const int wn   = tid >> 5;           // 0..3

    if (tid < 16) toks[tid] = tokens[(bt * 16 + tid) * T_ + t];
    __syncthreads();

    // Per-lane A row base pointer (gathered embedding row, bf16)
    const unsigned short* pa =
        emb_bf + (size_t)toks[lane & 15] * E_ + ((lane >> 4) << 3);

    v8f acc[8] = {};

    // Software-pipelined K loop: B frags 2 ahead, A frag 1 ahead.
    v16bf a_cur = load_afrag_g(pa);
    v16bf b_cur = load_bfrag(Wshuf, wn * 8 + 0, lane);
    v16bf b_nxt = load_bfrag(Wshuf, wn * 8 + 1, lane);
    for (int kt = 0; kt < 16; ++kt) {
        v16bf a_nxt = a_cur;
        if (kt < 15) a_nxt = load_afrag_g(pa + (kt + 1) * 32);
        const int fbase = kt * 32 + wn * 8;
#pragma unroll
        for (int nt = 0; nt < 8; ++nt) {
            v16bf b_pref = b_nxt;
            if (nt < 6)            b_pref = load_bfrag(Wshuf, fbase + nt + 2, lane);
            else if (kt < 15)      b_pref = load_bfrag(Wshuf, fbase + 32 + (nt - 6), lane);
            acc[nt] = __builtin_amdgcn_wmma_f32_16x16x32_bf16(
                false, a_cur, false, b_cur, (short)0, acc[nt], false, false);
            b_cur = b_nxt; b_nxt = b_pref;
        }
        a_cur = a_nxt;
    }

    const int nlo = lane & 15;
    const size_t fbase = (((size_t)t * NBT + bt) * 4 + wn) * 8;
#pragma unroll
    for (int nt = 0; nt < 8; ++nt) {
        float bb = bias[wn * 128 + nt * 16 + nlo];
        H8 o;
#pragma unroll
        for (int r = 0; r < 8; ++r) o.h[r] = f2bf(acc[nt][r] + bb);
        *(uint4*)(xk + (fbase + nt) * 256 + lane * 8) = o.q;
    }
}

// ---------------------------------------------------------------------------
// Kernel 2: RNN scan. WG owns 16 B-rows; H kept in LDS (bf16) for all 80 steps.
// h' = tanh(xk_t + H @ R); final: out[b] = sigmoid(h . fc_w + fc_b)
// ---------------------------------------------------------------------------
__global__ __launch_bounds__(128)
void rnn_kernel(const unsigned short* __restrict__ xk,
                const unsigned short* __restrict__ Rshuf,
                const float* __restrict__ fc_w, const float* __restrict__ fc_b,
                float* __restrict__ out) {
    __shared__ alignas(16) unsigned short Hlds[16 * STR];
    const int bt  = blockIdx.x;
    const int tid = threadIdx.x;
    const int lane = tid & 31;
    const int wn   = tid >> 5;

    for (int i = tid; i < (16 * STR) / 2; i += 128)
        ((unsigned int*)Hlds)[i] = 0u;        // h0 = 0
    __syncthreads();

    const int hi = (lane >> 4) & 1, nlo = lane & 15;

    for (int t = 0; t < T_; ++t) {
        v8f acc[8] = {};

        v16bf a_cur = load_afrag_lds(Hlds, 0, lane);
        v16bf b_cur = load_bfrag(Rshuf, wn * 8 + 0, lane);
        v16bf b_nxt = load_bfrag(Rshuf, wn * 8 + 1, lane);
        for (int kt = 0; kt < 16; ++kt) {
            v16bf a_nxt = a_cur;
            if (kt < 15) a_nxt = load_afrag_lds(Hlds, kt + 1, lane);
            const int fbase = kt * 32 + wn * 8;
#pragma unroll
            for (int nt = 0; nt < 8; ++nt) {
                v16bf b_pref = b_nxt;
                if (nt < 6)        b_pref = load_bfrag(Rshuf, fbase + nt + 2, lane);
                else if (kt < 15)  b_pref = load_bfrag(Rshuf, fbase + 32 + (nt - 6), lane);
                acc[nt] = __builtin_amdgcn_wmma_f32_16x16x32_bf16(
                    false, a_cur, false, b_cur, (short)0, acc[nt], false, false);
                b_cur = b_nxt; b_nxt = b_pref;
            }
            a_cur = a_nxt;
        }
        __syncthreads();                      // all waves done reading H

        const size_t fbase = (((size_t)t * NBT + bt) * 4 + wn) * 8;
#pragma unroll
        for (int nt = 0; nt < 8; ++nt) {
            H8 xin; xin.q = *(const uint4*)(xk + (fbase + nt) * 256 + lane * 8);
            int col = wn * 128 + nt * 16 + nlo;
#pragma unroll
            for (int r = 0; r < 8; ++r) {
                float v = acc[nt][r] + bf2f(xin.h[r]);
                Hlds[(hi * 8 + r) * STR + col] = f2bf(fast_tanh(v));
            }
        }
        __syncthreads();                      // H fully updated
    }

    if (tid < 16) {
        float sum = fc_b[0];
        for (int u = 0; u < U_; ++u)
            sum += bf2f(Hlds[tid * STR + u]) * fc_w[u];
        out[bt * 16 + tid] = 1.0f / (1.0f + expf(-sum));
    }
}

// ---------------------------------------------------------------------------
extern "C" void kernel_launch(void* const* d_in, const int* in_sizes, int n_in,
                              void* d_out, int out_size, void* d_ws, size_t ws_size,
                              hipStream_t stream) {
    const int*   tokens = (const int*)d_in[0];
    const float* emb    = (const float*)d_in[1];
    const float* kern   = (const float*)d_in[2];
    const float* rec    = (const float*)d_in[3];
    const float* bias   = (const float*)d_in[4];
    const float* fc_w   = (const float*)d_in[5];
    const float* fc_b   = (const float*)d_in[6];

    unsigned short* Wshuf  = (unsigned short*)d_ws;              // 512 KB
    unsigned short* Rshuf  = Wshuf + (size_t)E_ * U_;            // 512 KB
    unsigned short* emb_bf = Rshuf + (size_t)U_ * U_;            // 51.2 MB
    unsigned short* xk     = emb_bf + (size_t)V_ * E_;           // 80 MB (frag layout)

    shuffleB_kernel<<<512, 32, 0, stream>>>(kern, Wshuf);
    shuffleB_kernel<<<512, 32, 0, stream>>>(rec,  Rshuf);

    const int n4 = (V_ * E_) / 4;
    cvt_emb_kernel<<<(n4 + 255) / 256, 256, 0, stream>>>(emb, emb_bf, n4);

    dim3 g1(NBT, T_);
    proj_kernel<<<g1, 128, 0, stream>>>(tokens, emb_bf, Wshuf, bias, xk);

    rnn_kernel<<<NBT, 128, 0, stream>>>(xk, Rshuf, fc_w, fc_b, (float*)d_out);
}
